// AttnBlock_52261162057950
// MI455X (gfx1250) — matrix operand order repaired and checked by reading, not compiled, not run on hardware
//
#include <hip/hip_runtime.h>

// ---------------------------------------------------------------------------
// AttnBlock (VAE-style single-head attention) for MI455X / gfx1250.
// All heavy math runs through v_wmma_f32_16x16x32_bf16 (bf16 in, f32 acc).
// Tile data movement uses gfx1250 async loads (global_load_async_to_lds_b128,
// ASYNCcnt) with double-buffered LDS to overlap DMA with WMMA issue.
// Pipeline:
//   1. gn_stats + gn_apply      : GroupNorm -> hn bf16, layout (B, N, C)
//   2. f32->bf16 weight convert : wq/wk/wv/wp -> bf16
//   3. WMMA GEMM  q = hn*WqT+bq : (B,N,C) bf16
//                 k = hn*WkT+bk : (B,N,C) bf16
//                 v = hn*WvT+bv : stored transposed -> (B,C,N) bf16
//   4. WMMA GEMM  S = q*kT * a  : (B,N,N) f32   (alpha = C^-0.5)
//   5. softmax rows of S -> P   : (B,N,N) bf16
//   6. WMMA GEMM  O = P*vT      : (B,N,C) bf16
//   7. WMMA GEMM  out = O*WpT+bp + x : f32, transposed store -> (B,C,H,W)
// ---------------------------------------------------------------------------

typedef __attribute__((ext_vector_type(16))) __bf16 v16bf;
typedef __attribute__((ext_vector_type(8)))  __bf16 v8bf;
typedef __attribute__((ext_vector_type(8)))  float  v8f;

#define BATCH 4
#define CCH   512
#define NTOK  4096
#define NGRP  32
#define CPG   16          // CCH / NGRP

#define BM 256            // block tile M (4 waves x 64)
#define BN 128            // block tile N (2 waves x 64)
#define KT 32             // K per WMMA step
#define GEMM_THREADS 256  // 8 waves

static __device__ __forceinline__ unsigned short f2bf(float f) {
    union { float f; unsigned u; } x;
    x.f = f;
    unsigned r = x.u + 0x7FFFu + ((x.u >> 16) & 1u);   // round-to-nearest-even
    return (unsigned short)(r >> 16);
}

// ---------------------------------------------------------------------------
// GroupNorm: stats over (CPG * NTOK) contiguous floats per (b, g)
// ---------------------------------------------------------------------------
__global__ void gn_stats_kernel(const float* __restrict__ x, float* __restrict__ stats) {
    const int CHUNK = CPG * NTOK;  // 65536 contiguous elements
    const float* base = x + (size_t)blockIdx.x * CHUNK;
    __shared__ float s1[256];
    __shared__ float s2[256];
    float a = 0.f, b = 0.f;
    for (int i = threadIdx.x; i < CHUNK; i += 256) {
        float v = base[i];
        a += v;
        b += v * v;
    }
    s1[threadIdx.x] = a;
    s2[threadIdx.x] = b;
    __syncthreads();
    for (int s = 128; s > 0; s >>= 1) {
        if (threadIdx.x < s) {
            s1[threadIdx.x] += s1[threadIdx.x + s];
            s2[threadIdx.x] += s2[threadIdx.x + s];
        }
        __syncthreads();
    }
    if (threadIdx.x == 0) {
        float m = s1[0] / (float)CHUNK;
        float var = s2[0] / (float)CHUNK - m * m;
        stats[2 * blockIdx.x]     = m;
        stats[2 * blockIdx.x + 1] = rsqrtf(var + 1e-6f);
    }
}

// hn[b][n][c] = ((x[b][c][n]-mean)*rstd)*scale[c]+bias[c], bf16, (B,N,C) layout
__global__ void gn_apply_kernel(const float* __restrict__ x,
                                const float* __restrict__ stats,
                                const float* __restrict__ sc,
                                const float* __restrict__ bi,
                                unsigned short* __restrict__ hn) {
    int t = blockIdx.x * 256 + threadIdx.x;          // 0 .. B*N*C-1
    int c = t & (CCH - 1);
    int n = (t >> 9) & (NTOK - 1);
    int b = t >> 21;
    int g = c >> 4;                                   // c / CPG
    float m = stats[2 * (b * NGRP + g)];
    float r = stats[2 * (b * NGRP + g) + 1];
    float v = (x[((size_t)b * CCH + c) * NTOK + n] - m) * r * sc[c] + bi[c];
    hn[t] = f2bf(v);
}

__global__ void f32_to_bf16_kernel(const float* __restrict__ src,
                                   unsigned short* __restrict__ dst, int n) {
    int i = blockIdx.x * 256 + threadIdx.x;
    if (i < n) dst[i] = f2bf(src[i]);
}

// ---------------------------------------------------------------------------
// Row softmax: one block per row of NTOK f32 scores; emit bf16 probabilities.
// ---------------------------------------------------------------------------
__global__ void softmax_kernel(const float* __restrict__ S,
                               unsigned short* __restrict__ P) {
    __shared__ float red[256];
    const float* row = S + (size_t)blockIdx.x * NTOK;
    unsigned short* prow = P + (size_t)blockIdx.x * NTOK;
    int tid = threadIdx.x;

    float mx = -3.4e38f;
    for (int i = tid; i < NTOK; i += 256) mx = fmaxf(mx, row[i]);
    red[tid] = mx;
    __syncthreads();
    for (int s = 128; s > 0; s >>= 1) {
        if (tid < s) red[tid] = fmaxf(red[tid], red[tid + s]);
        __syncthreads();
    }
    mx = red[0];
    __syncthreads();

    float sum = 0.f;
    for (int i = tid; i < NTOK; i += 256) sum += __expf(row[i] - mx);
    red[tid] = sum;
    __syncthreads();
    for (int s = 128; s > 0; s >>= 1) {
        if (tid < s) red[tid] += red[tid + s];
        __syncthreads();
    }
    float inv = 1.0f / red[0];
    for (int i = tid; i < NTOK; i += 256) prow[i] = f2bf(__expf(row[i] - mx) * inv);
}

// ---------------------------------------------------------------------------
// Generic batched bf16 WMMA GEMM:
//   D[m][n] = alpha * sum_k A[m][k] * Bm[n][k]  (+ bias[n]) (+ resid)
// A   : row-major (lda), K contiguous
// Bm  : "column major" -- row n of Bm is column n of the math-B, K contiguous
// Out : bf16 or f32 (flags bit0), optionally transposed store out[n][m] (bit1)
// resid (only with transposed f32): resid[n * ldo + m] added.
//
// Block tile 256x128, K-step 32, 8 waves (4 M x 2 N), wave tile 64x64 =
// 4x4 WMMA fragments. K-tiles are double-buffered in LDS and fetched with
// gfx1250 async-to-LDS loads (ASYNCcnt), overlapped with WMMA of the
// current tile; one s_wait_asynccnt + one barrier per K-step.
// ---------------------------------------------------------------------------
__global__ __launch_bounds__(GEMM_THREADS)
void gemm_bf16_wmma_kernel(const unsigned short* __restrict__ A, int lda, long long strideA,
                           const unsigned short* __restrict__ Bm, int ldb, long long strideB,
                           int K,
                           const float* __restrict__ bias,
                           const float* __restrict__ resid, long long strideR,
                           void* __restrict__ Out, int ldo, long long strideO,
                           float alpha, int flags) {
    __shared__ __align__(128) unsigned short As[2][BM][KT];   // 2 x 16 KB
    __shared__ __align__(128) unsigned short Bs[2][BN][KT];   // 2 x  8 KB

    const int tid  = threadIdx.x;
    const int lane = tid & 31;
    const int wid  = tid >> 5;
    const int wm   = wid & 3;        // 4 waves along M (64 rows each)
    const int wn   = wid >> 2;       // 2 waves along N (64 cols each)

    const int m0 = blockIdx.x * BM;
    const int n0 = blockIdx.y * BN;
    const int bz = blockIdx.z;

    const unsigned short* Ab = A  + (size_t)bz * strideA;
    const unsigned short* Bb = Bm + (size_t)bz * strideB;
    const float* Rb = resid ? (resid + (size_t)bz * strideR) : nullptr;
    char* Ob = (char*)Out + (size_t)bz * strideO * ((flags & 1) ? 4 : 2);

    // fragment lane decomposition
    const int fl = lane & 15;                // row (A) / col (B,D)
    const int fh = lane >> 4;                // half selector

    // per-thread async-load slots:
    //   A: one thread per row (256 rows), 64 B per K-tile -> 4x b128
    //   B: two threads per row (128 rows), 32 B each      -> 2x b128
    const unsigned short* gA = Ab + (size_t)(m0 + tid) * lda;
    const unsigned short* gB = Bb + (size_t)(n0 + (tid >> 1)) * ldb + ((tid & 1) << 4);
    const unsigned ldsA[2] = { (unsigned)(uintptr_t)&As[0][tid][0],
                               (unsigned)(uintptr_t)&As[1][tid][0] };
    const unsigned ldsB[2] = { (unsigned)(uintptr_t)&Bs[0][tid >> 1][(tid & 1) << 4],
                               (unsigned)(uintptr_t)&Bs[1][tid >> 1][(tid & 1) << 4] };

    auto issue_tile = [&](int k0, int bsel) {
        unsigned long long a64 = (unsigned long long)(uintptr_t)(gA + k0);
        unsigned long long b64 = (unsigned long long)(uintptr_t)(gB + k0);
        // Instruction offset applies to both global and LDS addresses.
        asm volatile(
            "global_load_async_to_lds_b128 %0, %2, off\n\t"
            "global_load_async_to_lds_b128 %0, %2, off offset:16\n\t"
            "global_load_async_to_lds_b128 %0, %2, off offset:32\n\t"
            "global_load_async_to_lds_b128 %0, %2, off offset:48\n\t"
            "global_load_async_to_lds_b128 %1, %3, off\n\t"
            "global_load_async_to_lds_b128 %1, %3, off offset:16"
            :: "v"(ldsA[bsel]), "v"(ldsB[bsel]), "v"(a64), "v"(b64)
            : "memory");
    };

    v8f acc[4][4];
#pragma unroll
    for (int i = 0; i < 4; ++i)
#pragma unroll
        for (int j = 0; j < 4; ++j)
#pragma unroll
            for (int e = 0; e < 8; ++e) acc[i][j][e] = 0.f;

    issue_tile(0, 0);
    const int nk = K / KT;
    for (int t = 0; t < nk; ++t) {
        const int cur = t & 1;
        asm volatile("s_wait_asynccnt 0" ::: "memory");   // our tile landed in LDS
        __syncthreads();                                  // everyone's tile landed
        if (t + 1 < nk) issue_tile((t + 1) * KT, cur ^ 1);

        // A fragments: lane fl holds row M=fl; elems 0..7 = K(8*fh ..),
        // elems 8..15 = K(16+8*fh ..)  (ISA 16-bit A 16x32 layout)
        v16bf afr[4];
#pragma unroll
        for (int mf = 0; mf < 4; ++mf) {
            const unsigned short* rp = &As[cur][wm * 64 + mf * 16 + fl][0];
            v8bf lo = *(const v8bf*)(rp + 8 * fh);
            v8bf hi = *(const v8bf*)(rp + 16 + 8 * fh);
            v16bf a;
#pragma unroll
            for (int e = 0; e < 8; ++e) { a[e] = lo[e]; a[e + 8] = hi[e]; }
            afr[mf] = a;
        }
        // B fragments: lane (fl + 16*fh) holds column fl; elems 0..7 = K(8*fh ..),
        // elems 8..15 = K(16+8*fh ..)  (same K order as A)
        v16bf bfr[4];
#pragma unroll
        for (int nf = 0; nf < 4; ++nf) {
            const unsigned short* cp = &Bs[cur][wn * 64 + nf * 16 + fl][0];
            v8bf lo = *(const v8bf*)(cp + 8 * fh);
            v8bf hi = *(const v8bf*)(cp + 16 + 8 * fh);
            v16bf b;
#pragma unroll
            for (int e = 0; e < 8; ++e) { b[e] = lo[e]; b[e + 8] = hi[e]; }
            bfr[nf] = b;
        }

#pragma unroll
        for (int mf = 0; mf < 4; ++mf)
#pragma unroll
            for (int nf = 0; nf < 4; ++nf)
                acc[mf][nf] = __builtin_amdgcn_wmma_f32_16x16x32_bf16(
                    false, afr[mf], false, bfr[nf], (short)0, acc[mf][nf],
                    false, false);
    }

    // Epilogue. D layout: VGPR r, lane = fl + 16*fh -> (M = 8*fh + r, N = fl)
#pragma unroll
    for (int mf = 0; mf < 4; ++mf) {
#pragma unroll
        for (int nf = 0; nf < 4; ++nf) {
            int gr0 = m0 + wm * 64 + mf * 16 + 8 * fh;
            int gc  = n0 + wn * 64 + nf * 16 + fl;
            float bv = bias ? bias[gc] : 0.f;
#pragma unroll
            for (int r = 0; r < 8; ++r) {
                int gr = gr0 + r;
                float v = acc[mf][nf][r] * alpha + bv;
                if (flags & 2) {                       // transposed store out[n][m]
                    size_t idx = (size_t)gc * ldo + gr;
                    if (Rb) v += Rb[idx];
                    if (flags & 1) ((float*)Ob)[idx] = v;
                    else           ((unsigned short*)Ob)[idx] = f2bf(v);
                } else {                               // out[m][n]
                    size_t idx = (size_t)gr * ldo + gc;
                    if (flags & 1) ((float*)Ob)[idx] = v;
                    else           ((unsigned short*)Ob)[idx] = f2bf(v);
                }
            }
        }
    }
}

// ---------------------------------------------------------------------------
// Host-side orchestration
// ---------------------------------------------------------------------------
extern "C" void kernel_launch(void* const* d_in, const int* in_sizes, int n_in,
                              void* d_out, int out_size, void* d_ws, size_t ws_size,
                              hipStream_t stream) {
    (void)in_sizes; (void)n_in; (void)out_size; (void)ws_size;

    const float* x  = (const float*)d_in[0];
    const float* gs = (const float*)d_in[1];
    const float* gb = (const float*)d_in[2];
    const float* wq = (const float*)d_in[3];
    const float* bq = (const float*)d_in[4];
    const float* wk = (const float*)d_in[5];
    const float* bk = (const float*)d_in[6];
    const float* wv = (const float*)d_in[7];
    const float* bv = (const float*)d_in[8];
    const float* wp = (const float*)d_in[9];
    const float* bp = (const float*)d_in[10];

    const long long NC = (long long)NTOK * CCH;     // 2,097,152
    const long long NN = (long long)NTOK * NTOK;    // 16,777,216
    const long long CN = (long long)CCH * NTOK;

    // workspace carve-out (256B aligned slices)
    char* w = (char*)d_ws;
    auto carve = [&](size_t bytes) -> char* {
        char* p = w;
        w += (bytes + 255) & ~(size_t)255;
        return p;
    };
    float*          stats = (float*)         carve((size_t)BATCH * NGRP * 2 * 4);
    unsigned short* wqb   = (unsigned short*)carve((size_t)CCH * CCH * 2);
    unsigned short* wkb   = (unsigned short*)carve((size_t)CCH * CCH * 2);
    unsigned short* wvb   = (unsigned short*)carve((size_t)CCH * CCH * 2);
    unsigned short* wpb   = (unsigned short*)carve((size_t)CCH * CCH * 2);
    unsigned short* hn    = (unsigned short*)carve((size_t)BATCH * NC * 2);   // (B,N,C)
    unsigned short* qb    = (unsigned short*)carve((size_t)BATCH * NC * 2);   // (B,N,C)
    unsigned short* kb    = (unsigned short*)carve((size_t)BATCH * NC * 2);   // (B,N,C)
    unsigned short* vb    = (unsigned short*)carve((size_t)BATCH * NC * 2);   // (B,C,N)
    unsigned short* ob    = (unsigned short*)carve((size_t)BATCH * NC * 2);   // (B,N,C)
    float*          S     = (float*)         carve((size_t)BATCH * NN * 4);   // (B,N,N)
    unsigned short* P     = (unsigned short*)carve((size_t)BATCH * NN * 2);   // (B,N,N)

    const float attn_scale = 0.044194173824159216f;  // 512^-0.5

    // 1. GroupNorm
    gn_stats_kernel<<<dim3(BATCH * NGRP), dim3(256), 0, stream>>>(x, stats);
    gn_apply_kernel<<<dim3((BATCH * NTOK * CCH) / 256), dim3(256), 0, stream>>>(
        x, stats, gs, gb, hn);

    // 2. weights -> bf16
    f32_to_bf16_kernel<<<dim3((CCH * CCH) / 256), dim3(256), 0, stream>>>(wq, wqb, CCH * CCH);
    f32_to_bf16_kernel<<<dim3((CCH * CCH) / 256), dim3(256), 0, stream>>>(wk, wkb, CCH * CCH);
    f32_to_bf16_kernel<<<dim3((CCH * CCH) / 256), dim3(256), 0, stream>>>(wv, wvb, CCH * CCH);
    f32_to_bf16_kernel<<<dim3((CCH * CCH) / 256), dim3(256), 0, stream>>>(wp, wpb, CCH * CCH);

    dim3 blk(GEMM_THREADS);
    dim3 gProj(NTOK / BM, CCH / BN, BATCH);   // (16, 4, 4)
    dim3 gAttn(NTOK / BM, NTOK / BN, BATCH);  // (16, 32, 4)

    // 3. q / k / v projections
    gemm_bf16_wmma_kernel<<<gProj, blk, 0, stream>>>(
        hn, CCH, NC, wqb, CCH, 0, CCH, bq, nullptr, 0, qb, CCH, NC, 1.f, /*flags=*/0);
    gemm_bf16_wmma_kernel<<<gProj, blk, 0, stream>>>(
        hn, CCH, NC, wkb, CCH, 0, CCH, bk, nullptr, 0, kb, CCH, NC, 1.f, /*flags=*/0);
    gemm_bf16_wmma_kernel<<<gProj, blk, 0, stream>>>(   // v stored (B,C,N)
        hn, CCH, NC, wvb, CCH, 0, CCH, bv, nullptr, 0, vb, NTOK, CN, 1.f, /*flags=*/2);

    // 4. scores S = q @ k^T * scale   (f32 out)
    gemm_bf16_wmma_kernel<<<gAttn, blk, 0, stream>>>(
        qb, CCH, NC, kb, CCH, NC, CCH, nullptr, nullptr, 0, S, NTOK, NN,
        attn_scale, /*flags=*/1);

    // 5. softmax rows -> P (bf16)
    softmax_kernel<<<dim3(BATCH * NTOK), dim3(256), 0, stream>>>(S, P);

    // 6. O = P @ v^T   (B,N,C) bf16
    gemm_bf16_wmma_kernel<<<gProj, blk, 0, stream>>>(
        P, NTOK, NN, vb, NTOK, CN, NTOK, nullptr, nullptr, 0, ob, CCH, NC,
        1.f, /*flags=*/0);

    // 7. out = O @ wp^T + bp + x  -> f32, transposed store into (B,C,H,W)
    gemm_bf16_wmma_kernel<<<gProj, blk, 0, stream>>>(
        ob, CCH, NC, wpb, CCH, 0, CCH, bp, x, CN, d_out, NTOK, CN,
        1.f, /*flags=*/3);
}
